// DeformConv1d_13889924235372
// MI455X (gfx1250) — matrix operand order repaired and checked by reading, not compiled, not run on hardware
//
#include <hip/hip_runtime.h>

typedef __attribute__((ext_vector_type(4)))  float    v4f;
typedef __attribute__((ext_vector_type(8)))  float    v8f;
typedef __attribute__((ext_vector_type(16))) _Float16 v16h;

#define Bn    4
#define CIN   256
#define COUT  256
#define Ln    8192
#define Kn    3
#define DGn   4
#define CPER  64
#define OFFCH 12          // DG*K
#define RED   768         // CIN*K reduction length (r = k*256 + cin)
#define NT    32          // output-column tile of main kernel
#define RS    (RED + 8)   // padded LDS row stride in halves (776): bank-conflict-free

// ---------------------------------------------------------------------------
// Kernel 1: offset conv + mask conv (sigmoid). Plain fp32 VALU, weights in LDS.
// ---------------------------------------------------------------------------
__global__ void offmask_conv_kernel(const float* __restrict__ x,
                                    const float* __restrict__ w_off,
                                    const float* __restrict__ b_off,
                                    const float* __restrict__ w_mask,
                                    const float* __restrict__ b_mask,
                                    float* __restrict__ offs,
                                    float* __restrict__ msk) {
    __shared__ float wsm[OFFCH * CIN * Kn]; // 9216 floats = 36 KB
    const int tid = threadIdx.x;
    const int b   = blockIdx.x >> 5;          // 32 l-tiles per batch
    const int l   = ((blockIdx.x & 31) << 8) + tid;

    for (int pass = 0; pass < 2; ++pass) {
        const float* w = (pass == 0) ? w_off : w_mask;
        for (int i = tid; i < OFFCH * CIN * Kn; i += 256) wsm[i] = w[i];
        __syncthreads();

        float acc[OFFCH];
        #pragma unroll
        for (int d = 0; d < OFFCH; ++d) acc[d] = 0.0f;

        const float* xb = x + (size_t)b * CIN * Ln;
        for (int cin = 0; cin < CIN; ++cin) {
            const float* xc = xb + (size_t)cin * Ln;
            float xm1 = (l >= 1)      ? xc[l - 1] : 0.0f;
            float x0v =                 xc[l];
            float xp1 = (l + 1 < Ln)  ? xc[l + 1] : 0.0f;
            const float* wc = &wsm[cin * Kn];
            #pragma unroll
            for (int dk = 0; dk < OFFCH; ++dk) {
                const float* wp = wc + dk * (CIN * Kn);
                acc[dk] = fmaf(wp[0], xm1, fmaf(wp[1], x0v, fmaf(wp[2], xp1, acc[dk])));
            }
        }
        if (pass == 0) {
            #pragma unroll
            for (int dk = 0; dk < OFFCH; ++dk)
                offs[((size_t)(b * OFFCH + dk) << 13) + l] = acc[dk] + b_off[dk];
        } else {
            #pragma unroll
            for (int dk = 0; dk < OFFCH; ++dk) {
                float v = acc[dk] + b_mask[dk];
                msk[((size_t)(b * OFFCH + dk) << 13) + l] = 1.0f / (1.0f + expf(-v));
            }
        }
        __syncthreads();
    }
}

// ---------------------------------------------------------------------------
// Kernel 2: transpose+convert x [B,C,L] f32 -> xT [B,L,C] f16 (LDS-tiled).
// ---------------------------------------------------------------------------
__global__ void transpose_x_kernel(const float* __restrict__ x,
                                   _Float16* __restrict__ xT) {
    __shared__ float tile[64][65];
    const int tid = threadIdx.x;
    const int bid = blockIdx.x;               // b*(128*4) + lt*4 + ct
    const int ct  = bid & 3;
    const int lt  = (bid >> 2) & 127;
    const int b   = bid >> 9;
    const int c0  = ct << 6, l0 = lt << 6;

    #pragma unroll
    for (int it = 0; it < 16; ++it) {
        int idx = it * 256 + tid;
        int cl = idx >> 6, ll = idx & 63;     // lanes contiguous in l -> coalesced
        tile[cl][ll] = x[((size_t)(b * CIN + c0 + cl) << 13) + l0 + ll];
    }
    __syncthreads();
    #pragma unroll
    for (int it = 0; it < 16; ++it) {
        int idx = it * 256 + tid;
        int ll = idx >> 6, cl = idx & 63;     // lanes contiguous in c -> coalesced
        xT[((size_t)(b * Ln + l0 + ll) << 8) + c0 + cl] = (_Float16)tile[cl][ll];
    }
}

// ---------------------------------------------------------------------------
// Kernel 3: repack weight [COUT,CIN,K] f32 -> Wf [COUT][RED] f16, r = k*256+cin.
// ---------------------------------------------------------------------------
__global__ void wconvert_kernel(const float* __restrict__ w,
                                _Float16* __restrict__ Wf) {
    int idx = blockIdx.x * 256 + threadIdx.x; // COUT*RED = 196608 total
    int o = idx / RED, r = idx % RED;
    int k = r >> 8, cin = r & 255;
    Wf[idx] = (_Float16)w[(size_t)(o * CIN + cin) * Kn + k];
}

// ---------------------------------------------------------------------------
// Kernel 4: gather/interp/modulate S-tile into LDS, then WMMA GEMM.
// ---------------------------------------------------------------------------
__device__ inline v16h load_fragA(const _Float16* p) {   // halves [0..7],[16..23]
    v16h r;
    v4f* rp = reinterpret_cast<v4f*>(&r);
    rp[0] = *reinterpret_cast<const v4f*>(p);
    rp[1] = *reinterpret_cast<const v4f*>(p + 16);
    return r;
}
__device__ inline v16h load_fragB(const _Float16* p) {   // halves [0..7],[16..23]
    v16h r;
    v4f* rp = reinterpret_cast<v4f*>(&r);
    rp[0] = *reinterpret_cast<const v4f*>(p);
    rp[1] = *reinterpret_cast<const v4f*>(p + 16);
    return r;
}

__global__ void __launch_bounds__(256)
deform_gemm_kernel(const _Float16* __restrict__ xT,
                   const _Float16* __restrict__ Wf,
                   const float* __restrict__ offs,
                   const float* __restrict__ msk,
                   const float* __restrict__ bias,
                   float* __restrict__ out) {
    __shared__ __align__(16) _Float16 sT[NT * RS]; // 32 x 776 halves = 48.5 KB

    const int tid = threadIdx.x;
    const int b   = blockIdx.x >> 8;           // 256 column-tiles per batch
    const int l0  = (blockIdx.x & 255) << 5;

    // ---- Phase 1: build S^T tile [NT rows][RED] in LDS (f16) -------------
    // 384 tasks = (l_local, d, k); 64 lanes per task cover the 64 channels.
    {
        const int lane_c = tid & 63;
        const int group  = tid >> 6;           // 0..3
        for (int task = group; task < NT * OFFCH; task += 4) {
            const int l_local = task / OFFCH;
            const int dk      = task % OFFCH;  // d*3 + k, matches offset channel
            const int d = dk / Kn, k = dk % Kn;
            const int l = l0 + l_local;

            const size_t oidx = ((size_t)(b * OFFCH + dk) << 13) + l;
            const float off = offs[oidx];
            const float mm  = msk[oidx];

            const float p  = (float)(l - 1 + k) + off;
            const float p0 = floorf(p);
            const float t  = p - p0;
            const int  i0  = (int)p0;
            const int  i1  = i0 + 1;
            const int  ch  = (d << 6) + lane_c;

            const int idx0 = min(max(i0, 0), Ln - 1);
            const int idx1 = min(max(i1, 0), Ln - 1);
            const _Float16* xb = xT + ((size_t)b << 21); // b*L*256
            float v0 = (i0 >= 0 && i0 < Ln) ? (float)xb[((size_t)idx0 << 8) + ch] : 0.0f;
            float v1 = (i1 >= 0 && i1 < Ln) ? (float)xb[((size_t)idx1 << 8) + ch] : 0.0f;

            float s = fmaf(v1 - v0, t, v0) * mm;
            sT[l_local * RS + (k << 8) + ch] = (_Float16)s;
        }
    }
    __syncthreads();

    // ---- Phase 2: C[256 x 32] = Wf[256 x 768] * S[768 x 32] via WMMA -----
    const int wave = tid >> 5;                 // 8 waves
    const int lane = tid & 31;
    const int hl   = lane & 15;
    const int kselA = (lane < 16) ? 0 : 8;     // A-frag K sub-offset (halves)
    const int kselB = (lane < 16) ? 0 : 8;     // B-frag K sub-offset (halves)

    #pragma unroll
    for (int mi = 0; mi < 2; ++mi) {
        const int Mbase = (wave * 2 + mi) << 4;
        const _Float16* aRow = Wf + (size_t)(Mbase + hl) * RED;
        const _Float16* bRow0 = sT + hl * RS;          // N-tile 0: cols 0..15
        const _Float16* bRow1 = sT + (16 + hl) * RS;   // N-tile 1: cols 16..31

        v8f acc0 = {0.f, 0.f, 0.f, 0.f, 0.f, 0.f, 0.f, 0.f};
        v8f acc1 = {0.f, 0.f, 0.f, 0.f, 0.f, 0.f, 0.f, 0.f};

        for (int kk = 0; kk < RED / 32; ++kk) {
            const int kb = kk << 5;
            v16h a  = load_fragA(aRow  + kb + kselA);
            v16h b0 = load_fragB(bRow0 + kb + kselB);
            v16h b1 = load_fragB(bRow1 + kb + kselB);
            acc0 = __builtin_amdgcn_wmma_f32_16x16x32_f16(
                false, a, false, b0, (short)0, acc0, false, false);
            acc1 = __builtin_amdgcn_wmma_f32_16x16x32_f16(
                false, a, false, b1, (short)0, acc1, false, false);
        }

        // C/D layout: VGPR j -> M = Mbase + j (lanes 0-15) or Mbase + 8 + j
        const int mofs = (lane < 16) ? 0 : 8;
        #pragma unroll
        for (int j = 0; j < 8; ++j) {
            const int oc  = Mbase + mofs + j;
            const float bv = bias[oc];
            const size_t rowo = ((size_t)(b * COUT + oc) << 13);
            out[rowo + l0 + hl]      = acc0[j] + bv;
            out[rowo + l0 + 16 + hl] = acc1[j] + bv;
        }
    }
}

// ---------------------------------------------------------------------------
extern "C" void kernel_launch(void* const* d_in, const int* in_sizes, int n_in,
                              void* d_out, int out_size, void* d_ws, size_t ws_size,
                              hipStream_t stream) {
    const float* x      = (const float*)d_in[0];
    const float* w_off  = (const float*)d_in[1];
    const float* b_off  = (const float*)d_in[2];
    const float* w_mask = (const float*)d_in[3];
    const float* b_mask = (const float*)d_in[4];
    const float* weight = (const float*)d_in[5];
    const float* bias   = (const float*)d_in[6];
    float* out = (float*)d_out;

    // Workspace layout (bytes)
    char* ws = (char*)d_ws;
    _Float16* xT   = (_Float16*)(ws);                       // 16,777,216 B
    _Float16* Wf   = (_Float16*)(ws + 16777216);            //    393,216 B
    float*    offs = (float*)   (ws + 16777216 + 393216);   //  1,572,864 B
    float*    msk  = (float*)   (ws + 16777216 + 393216 + 1572864);

    offmask_conv_kernel<<<Bn * (Ln / 256), 256, 0, stream>>>(
        x, w_off, b_off, w_mask, b_mask, offs, msk);
    transpose_x_kernel<<<Bn * (Ln / 64) * (CIN / 64), 256, 0, stream>>>(x, xT);
    wconvert_kernel<<<(COUT * RED) / 256, 256, 0, stream>>>(weight, Wf);
    deform_gemm_kernel<<<Bn * (Ln / NT), 256, 0, stream>>>(
        xT, Wf, offs, msk, bias, out);
}
